// Kmeans_89472758710293
// MI455X (gfx1250) — compile-verified, matrix operand order checked by reading, not run on hardware
//
#include <hip/hip_runtime.h>
#include <math.h>

// Problem constants (from setup_inputs)
#define N_PTS 8192
#define K_CLU 8192
#define F_DIM 4096
#define LRATE 1e-4f
#define EPSI  1e-8f

typedef __attribute__((ext_vector_type(16))) __bf16 v16bf;
typedef __attribute__((ext_vector_type(8)))  __bf16 v8bf;
typedef __attribute__((ext_vector_type(8)))  float  v8f;
typedef __attribute__((ext_vector_type(4)))  float  v4f;

union FragU { v16bf v; v8bf h[2]; };

// ---------------- workspace layout (float offsets) ----------------
#define WS_SUMCN    0
#define WS_SUMCS2   1
#define WS_PLACC    2
#define WS_SUMDELTA 3
#define WS_DELTAK   4
#define WS_GBASE    5
#define WS_CN       8
#define WS_XN       (WS_CN + K_CLU)
#define WS_XDC      (WS_XN + N_PTS)
#define WS_DELTA    (WS_XDC + N_PTS)
#define WS_UPD      (WS_DELTA + K_CLU)
#define WS_COUNTS   (WS_UPD + K_CLU)
#define WS_CSUM     (WS_COUNTS + K_CLU)
#define WS_NEAREST  (WS_CSUM + F_DIM)   // ints

// ---------------- generic zero ----------------
__global__ void k_zero(float* p, size_t n) {
    size_t i = (size_t)blockIdx.x * blockDim.x + threadIdx.x;
    size_t stride = (size_t)gridDim.x * blockDim.x;
    for (; i < n; i += stride) p[i] = 0.0f;
}

// ---------------- centroid row norms: cn[k], sumCn ----------------
__global__ void k_centroid_rownorm(const float* __restrict__ c,
                                   float* __restrict__ cn,
                                   float* __restrict__ sumCn) {
    __shared__ float red[256];
    int k = blockIdx.x;
    const float* row = c + (size_t)k * F_DIM;
    float acc = 0.0f;
    for (int f = threadIdx.x; f < F_DIM; f += 256) { float v = row[f]; acc += v * v; }
    red[threadIdx.x] = acc; __syncthreads();
    for (int s = 128; s > 0; s >>= 1) {
        if ((int)threadIdx.x < s) red[threadIdx.x] += red[threadIdx.x + s];
        __syncthreads();
    }
    if (threadIdx.x == 0) { cn[k] = red[0]; atomicAdd(sumCn, red[0]); }
}

// ---------------- centroid column sums: csum[f], sumCsum2 ----------------
__global__ void k_centroid_colsum(const float* __restrict__ c,
                                  float* __restrict__ csum,
                                  float* __restrict__ sumCs2) {
    __shared__ float red[256];
    int f = blockIdx.x * 256 + threadIdx.x;
    float acc = 0.0f;
    for (int k = 0; k < K_CLU; ++k) acc += c[(size_t)k * F_DIM + f];
    csum[f] = acc;
    red[threadIdx.x] = acc * acc; __syncthreads();
    for (int s = 128; s > 0; s >>= 1) {
        if ((int)threadIdx.x < s) red[threadIdx.x] += red[threadIdx.x + s];
        __syncthreads();
    }
    if (threadIdx.x == 0) atomicAdd(sumCs2, red[0]);
}

// ---------------- x stats: xn[n] = sum x^2, xdc[n] = x . csum ----------------
__global__ void k_x_stats(const float* __restrict__ x, const float* __restrict__ csum,
                          float* __restrict__ xn, float* __restrict__ xdc) {
    __shared__ float r0[256];
    __shared__ float r1[256];
    int n = blockIdx.x;
    const float* row = x + (size_t)n * F_DIM;
    float a = 0.0f, b = 0.0f;
    for (int f = threadIdx.x; f < F_DIM; f += 256) {
        float v = row[f];
        a += v * v;
        b += v * csum[f];
    }
    r0[threadIdx.x] = a; r1[threadIdx.x] = b; __syncthreads();
    for (int s = 128; s > 0; s >>= 1) {
        if ((int)threadIdx.x < s) { r0[threadIdx.x] += r0[threadIdx.x + s];
                                    r1[threadIdx.x] += r1[threadIdx.x + s]; }
        __syncthreads();
    }
    if (threadIdx.x == 0) { xn[n] = r0[0]; xdc[n] = r1[0]; }
}

// ---------------- main WMMA argmin kernel ----------------
// WG = 256 threads (8 waves), owns 64 points (n), sweeps all K centroids in
// tiles of 128, F in steps of 32.  fp32 -> bf16 hi/lo split, 3 WMMAs per tile.
// Running per-point minimum of (cn[k] - 2*dot) kept as packed u64 key via
// ds_min_u64 atomics in LDS.
#define APITCH 40   // bf16 elems per LDS row (32 data + 8 pad -> conflict-free b128)

__global__ void __launch_bounds__(256) k_argmin(const float* __restrict__ x,
                                                const float* __restrict__ cen,
                                                const float* __restrict__ cn,
                                                int* __restrict__ nearest) {
    __shared__ __attribute__((aligned(16))) __bf16 As_hi[128 * APITCH];
    __shared__ __attribute__((aligned(16))) __bf16 As_lo[128 * APITCH];
    __shared__ __attribute__((aligned(16))) __bf16 Bs_hi[64 * APITCH];
    __shared__ __attribute__((aligned(16))) __bf16 Bs_lo[64 * APITCH];
    __shared__ unsigned long long best[64];

    const int tid  = threadIdx.x;
    const int lane = tid & 31;
    const int wave = tid >> 5;          // 0..7, each wave owns a 16-row k strip
    const int nbase = blockIdx.x * 64;
    const int hiHalf = (lane >= 16) ? 1 : 0;
    const int lrow = lane & 15;

    if (tid < 64) best[tid] = 0xFFFFFFFFFFFFFFFFull;
    __syncthreads();

    const int f4 = (tid & 7) * 4;       // float4 column within 32-wide f tile
    const int r0 = tid >> 3;            // base row for tile loads

    for (int kbase = 0; kbase < K_CLU; kbase += 128) {
        v8f acc0 = {}, acc1 = {}, acc2 = {}, acc3 = {};

        for (int fb = 0; fb < F_DIM; fb += 32) {
            __syncthreads();
            // ---- stage A tile (centroids 128x32) with fp32->bf16 hi/lo split
            #pragma unroll
            for (int it = 0; it < 4; ++it) {
                int r = r0 + 32 * it;
                v4f vv = *(const v4f*)(cen + (size_t)(kbase + r) * F_DIM + fb + f4);
                #pragma unroll
                for (int j = 0; j < 4; ++j) {
                    float fv = vv[j];
                    __bf16 h = (__bf16)fv;
                    __bf16 l = (__bf16)(fv - (float)h);
                    As_hi[r * APITCH + f4 + j] = h;
                    As_lo[r * APITCH + f4 + j] = l;
                }
            }
            // ---- stage B tile (x 64x32), B column n == x row n
            #pragma unroll
            for (int it = 0; it < 2; ++it) {
                int r = r0 + 32 * it;
                v4f vv = *(const v4f*)(x + (size_t)(nbase + r) * F_DIM + fb + f4);
                #pragma unroll
                for (int j = 0; j < 4; ++j) {
                    float fv = vv[j];
                    __bf16 h = (__bf16)fv;
                    __bf16 l = (__bf16)(fv - (float)h);
                    Bs_hi[r * APITCH + f4 + j] = h;
                    Bs_lo[r * APITCH + f4 + j] = l;
                }
            }
            __syncthreads();

            // ---- A fragment for this wave's 16-row strip (2 x b128 per frag)
            int aoff = (wave * 16 + lrow) * APITCH + hiHalf * 8;
            FragU ah, al;
            ah.h[0] = *(const v8bf*)&As_hi[aoff]; ah.h[1] = *(const v8bf*)&As_hi[aoff + 16];
            al.h[0] = *(const v8bf*)&As_lo[aoff]; al.h[1] = *(const v8bf*)&As_lo[aoff + 16];

            #pragma unroll
            for (int s = 0; s < 4; ++s) {
                int boff = (s * 16 + lrow) * APITCH + hiHalf * 8;
                FragU bh, bl;
                bh.h[0] = *(const v8bf*)&Bs_hi[boff]; bh.h[1] = *(const v8bf*)&Bs_hi[boff + 16];
                bl.h[0] = *(const v8bf*)&Bs_lo[boff]; bl.h[1] = *(const v8bf*)&Bs_lo[boff + 16];
                v8f* ap = (s == 0) ? &acc0 : (s == 1) ? &acc1 : (s == 2) ? &acc2 : &acc3;
                // hi*hi + hi*lo + lo*hi  ~= fp32 dot
                *ap = __builtin_amdgcn_wmma_f32_16x16x32_bf16(false, ah.v, false, bh.v, (short)0, *ap, false, false);
                *ap = __builtin_amdgcn_wmma_f32_16x16x32_bf16(false, ah.v, false, bl.v, (short)0, *ap, false, false);
                *ap = __builtin_amdgcn_wmma_f32_16x16x32_bf16(false, al.v, false, bh.v, (short)0, *ap, false, false);
            }
        }

        // ---- fold this k-tile into running per-point minimum
        // C layout: lane = n (col) in 0..15 (both halves), VGPR r -> m = r + 8*hiHalf
        #pragma unroll
        for (int s = 0; s < 4; ++s) {
            v8f a = (s == 0) ? acc0 : (s == 1) ? acc1 : (s == 2) ? acc2 : acc3;
            float bestS = 3.4e38f;
            int   bestK = 0;
            #pragma unroll
            for (int r = 0; r < 8; ++r) {
                int kk = kbase + wave * 16 + hiHalf * 8 + r;
                float sc = cn[kk] - 2.0f * a[r];   // xn[n] constant per column
                if (sc < bestS) { bestS = sc; bestK = kk; }
            }
            unsigned int mb = __float_as_uint(bestS);
            mb ^= (unsigned int)((int)mb >> 31) | 0x80000000u;   // order-preserving map
            unsigned long long key = ((unsigned long long)mb << 32) | (unsigned int)bestK;
            atomicMin(&best[s * 16 + lrow], key);                // ds_min_u64
        }
    }
    __syncthreads();
    if (tid < 64) nearest[nbase + tid] = (int)(best[tid] & 0xFFFFFFFFu);
}

// ---------------- segment count + segment sums (into d_out centroid slot) ---
__global__ void k_segment(const float* __restrict__ x, const int* __restrict__ nearest,
                          float* __restrict__ counts, float* __restrict__ sums) {
    __shared__ int kk;
    int n = blockIdx.x;
    if (threadIdx.x == 0) { kk = nearest[n]; atomicAdd(&counts[kk], 1.0f); }
    __syncthreads();
    const float* row = x + (size_t)n * F_DIM;
    float* srow = sums + (size_t)kk * F_DIM;
    for (int f = threadIdx.x; f < F_DIM; f += 256) atomicAdd(&srow[f], row[f]);
}

// ---------------- point_loss gather: sum colsum[nearest[n]] ----------------
__global__ void k_pointloss(const int* __restrict__ nearest, const float* __restrict__ xn,
                            const float* __restrict__ xdc, const float* __restrict__ sumCn,
                            float* __restrict__ plAcc) {
    __shared__ float red[256];
    int n = blockIdx.x * 256 + threadIdx.x;
    int j = nearest[n];
    float val = *sumCn + (float)K_CLU * xn[j] - 2.0f * xdc[j];
    red[threadIdx.x] = val; __syncthreads();
    for (int s = 128; s > 0; s >>= 1) {
        if ((int)threadIdx.x < s) red[threadIdx.x] += red[threadIdx.x + s];
        __syncthreads();
    }
    if (threadIdx.x == 0) atomicAdd(plAcc, red[0]);
}

// ---------------- per-cluster delta ----------------
__global__ void k_delta(const float* __restrict__ cen, const float* __restrict__ sums,
                        const float* __restrict__ counts, float* __restrict__ delta,
                        float* __restrict__ sumDelta, float* __restrict__ deltaK) {
    __shared__ float red[256];
    int k = blockIdx.x;
    float cnt = counts[k];
    float inv = 1.0f / fmaxf(cnt, 1.0f);
    const float* crow = cen + (size_t)k * F_DIM;
    const float* srow = sums + (size_t)k * F_DIM;
    float acc = 0.0f;
    for (int f = threadIdx.x; f < F_DIM; f += 256) {
        float d = srow[f] * inv - crow[f];
        acc += d * d;
    }
    red[threadIdx.x] = acc; __syncthreads();
    for (int s = 128; s > 0; s >>= 1) {
        if ((int)threadIdx.x < s) red[threadIdx.x] += red[threadIdx.x + s];
        __syncthreads();
    }
    if (threadIdx.x == 0) {
        bool ne = cnt > 0.0f;
        float d = ne ? red[0] : 0.0f;
        delta[k] = d;
        if (ne) { atomicAdd(sumDelta, d); atomicAdd(deltaK, 1.0f); }
    }
}

// ---------------- finalize scalars ----------------
__global__ void k_finalize(const float* __restrict__ ws, const float* __restrict__ betas,
                           float* __restrict__ out, float* __restrict__ gBase) {
    float pl = ws[WS_PLACC] / (float)N_PTS;
    float cl = 1.0f / (2.0f * (float)K_CLU * ws[WS_SUMCN] - 2.0f * ws[WS_SUMCS2]);
    out[0] = ws[WS_SUMDELTA] / ws[WS_DELTAK];
    out[1] = pl;
    out[2] = cl;
    *gBase = betas[2] * pl + betas[3] * cl;
}

// ---------------- Adam state update ----------------
__global__ void k_adam(const float* __restrict__ m, const float* __restrict__ v,
                       const float* __restrict__ betas, const int* __restrict__ step,
                       const float* __restrict__ delta, const float* __restrict__ counts,
                       const float* __restrict__ gBase, float* __restrict__ upd,
                       float* __restrict__ mOut, float* __restrict__ vOut) {
    int k = blockIdx.x * 256 + threadIdx.x;
    float b0 = betas[0], b1 = betas[1];
    bool ne = counts[k] > 0.0f;
    float g = delta[k] + *gBase;
    float mn = ne ? (b0 * m[k] + (1.0f - b0) * g)     : m[k];
    float vn = ne ? (b1 * v[k] + (1.0f - b1) * g * g) : v[k];
    float st = (float)(*step);
    float mh = mn / (1.0f - powf(b0, st));
    float vh = vn / (1.0f - powf(b1, st));
    float u = LRATE * mh / (sqrtf(vh) + EPSI);
    upd[k] = ne ? u : 0.0f;
    mOut[k] = mn;
    vOut[k] = vn;
}

// ---------------- write next centroids (overwrites sums region) ------------
__global__ void k_update_centroids(const float* __restrict__ cen, const float* __restrict__ upd,
                                   float* __restrict__ outC) {
    size_t i = (size_t)blockIdx.x * blockDim.x + threadIdx.x;
    size_t stride = (size_t)gridDim.x * blockDim.x;
    size_t total = (size_t)K_CLU * F_DIM;
    for (; i < total; i += stride) {
        int k = (int)(i >> 12);   // F_DIM == 4096
        outC[i] = cen[i] - upd[k];
    }
}

extern "C" void kernel_launch(void* const* d_in, const int* in_sizes, int n_in,
                              void* d_out, int out_size, void* d_ws, size_t ws_size,
                              hipStream_t stream) {
    (void)in_sizes; (void)n_in; (void)out_size; (void)ws_size;
    const float* x     = (const float*)d_in[0];
    const float* cen   = (const float*)d_in[1];
    const float* m     = (const float*)d_in[2];
    const float* v     = (const float*)d_in[3];
    const float* betas = (const float*)d_in[4];
    const int*   step  = (const int*)d_in[5];

    float* out   = (float*)d_out;
    float* outC  = out + 3;
    float* outM  = out + 3 + (size_t)K_CLU * F_DIM;
    float* outV  = outM + K_CLU;

    float* ws      = (float*)d_ws;
    float* cn      = ws + WS_CN;
    float* xn      = ws + WS_XN;
    float* xdc     = ws + WS_XDC;
    float* delta   = ws + WS_DELTA;
    float* upd     = ws + WS_UPD;
    float* counts  = ws + WS_COUNTS;
    float* csum    = ws + WS_CSUM;
    int*   nearest = (int*)(ws + WS_NEAREST);

    // zero accumulators (scalars + counts) and segment-sum region of d_out
    k_zero<<<1, 256, 0, stream>>>(ws, 8);
    k_zero<<<32, 256, 0, stream>>>(counts, K_CLU);
    k_zero<<<4096, 256, 0, stream>>>(outC, (size_t)K_CLU * F_DIM);

    // cheap stats
    k_centroid_rownorm<<<K_CLU, 256, 0, stream>>>(cen, cn, ws + WS_SUMCN);
    k_centroid_colsum<<<F_DIM / 256, 256, 0, stream>>>(cen, csum, ws + WS_SUMCS2);
    k_x_stats<<<N_PTS, 256, 0, stream>>>(x, csum, xn, xdc);

    // dominant WMMA GEMM + fused argmin
    k_argmin<<<N_PTS / 64, 256, 0, stream>>>(x, cen, cn, nearest);

    // segment reductions + losses
    k_segment<<<N_PTS, 256, 0, stream>>>(x, nearest, counts, outC);
    k_pointloss<<<N_PTS / 256, 256, 0, stream>>>(nearest, xn, xdc, ws + WS_SUMCN, ws + WS_PLACC);
    k_delta<<<K_CLU, 256, 0, stream>>>(cen, outC, counts, delta, ws + WS_SUMDELTA, ws + WS_DELTAK);
    k_finalize<<<1, 1, 0, stream>>>(ws, betas, out, ws + WS_GBASE);

    // Adam + centroid update (overwrites sums region with next_centroids)
    k_adam<<<K_CLU / 256, 256, 0, stream>>>(m, v, betas, step, delta, counts,
                                            ws + WS_GBASE, upd, outM, outV);
    k_update_centroids<<<4096, 256, 0, stream>>>(cen, upd, outC);
}
